// BertSelfAttention_25074019074436
// MI455X (gfx1250) — compile-verified, matrix operand order checked by reading, not run on hardware
//
#include <hip/hip_runtime.h>
#include <hip/hip_bf16.h>
#include <cstdint>
#include <cstddef>

// ---------------------------------------------------------------------------
// BERT self-attention for gfx1250 (MI455X), wave32 + WMMA bf16 + async-to-LDS.
//   B=4, S=2048, HID=1024, NH=16, HD=64.  fp32 in/out, bf16 matrix cores.
//
// Pipeline:
//   0) one-shot converts: X -> bf16 (same layout), W -> bf16 TRANSPOSED [n][k]
//   1) QKV GEMM (64x64 tile / 4 waves): all staging = async global->LDS b128,
//      double-buffered so DMA overlaps WMMA. Q,K stored [B,NH,S,HD]; V stored
//      TRANSPOSED [B,NH,HD,S] so attention never transposes in LDS.
//   2) flash attention: double-buffered async K/V staging, WMMA bf16 for
//      QK^T and P*V, online softmax via 16-lane shfl reductions.
// ---------------------------------------------------------------------------

#define B_   4
#define S_   2048
#define HID_ 1024
#define NH_  16
#define HD_  64

typedef __attribute__((ext_vector_type(16))) __bf16 v16bf;
typedef __attribute__((ext_vector_type(8)))  float  v8f;

// fp32 -> bf16, round-to-nearest-even (scalar fallback)
__device__ __forceinline__ unsigned short f2bf(float f) {
  unsigned int u = __float_as_uint(f);
  u += 0x7FFFu + ((u >> 16) & 1u);
  return (unsigned short)(u >> 16);
}

// Pack two fp32 -> packed bf16x2 (one v_cvt_pk_bf16_f32 when available)
#if __has_builtin(__builtin_amdgcn_cvt_pk_bf16_f32)
typedef __attribute__((ext_vector_type(2))) __bf16 v2bf;
__device__ __forceinline__ unsigned pack2bf(float a, float b) {
  union { v2bf v; unsigned u; } cv;
  cv.v = __builtin_amdgcn_cvt_pk_bf16_f32(a, b);
  return cv.u;
}
#else
__device__ __forceinline__ unsigned pack2bf(float a, float b) {
  return (unsigned)f2bf(a) | ((unsigned)f2bf(b) << 16);
}
#endif

// Async global->LDS 16-byte copy (per-lane). Generic pointers to __shared__
// carry the LDS byte offset in their low 32 bits (ISA 10.2 aperture rules).
__device__ __forceinline__ void async_copy_b128(void* lds, const void* gaddr) {
  unsigned ldsOff = (unsigned)(size_t)lds;
  asm volatile("global_load_async_to_lds_b128 %0, %1, off"
               :: "v"(ldsOff), "v"(gaddr) : "memory");
}
__device__ __forceinline__ void wait_async0() {
  asm volatile("s_wait_asynccnt 0x0" ::: "memory");
}

union FragU { uint4 q[2]; v16bf v; };

// Load a 16x32 bf16 A/B fragment from a row-major LDS tile.
// ISA 7.12.2 (16-bit A 16x32): lane holds row = lane%16; lanes 0-15 hold
// k {0..7, 16..23}, lanes 16-31 hold k {8..15, 24..31}; two bf16 per VGPR
// => two contiguous 16-byte chunks at k=kb and k=kb+16.
__device__ __forceinline__ v16bf lds_frag(const unsigned short* base,
                                          int rowBase, int stride, int kOff) {
  const int lane = threadIdx.x & 31;
  const int r    = lane & 15;
  const int kb   = (lane >> 4) << 3;              // 0 or 8
  const unsigned short* p = base + (rowBase + r) * stride + kOff + kb;
  FragU u;
  u.q[0] = *(const uint4*)(p);
  u.q[1] = *(const uint4*)(p + 16);
  return u.v;
}

// ---------------------------------------------------------------------------
// Kernel 0a: fp32 X -> bf16 X (same layout), vectorized 4-wide
// ---------------------------------------------------------------------------
__global__ void convert_x_kernel(const float* __restrict__ src,
                                 unsigned short* __restrict__ dst) {
  int i = blockIdx.x * 256 + threadIdx.x;     // float4 chunk id
  float4 f = ((const float4*)src)[i];
  uint2 pk;
  pk.x = pack2bf(f.x, f.y);
  pk.y = pack2bf(f.z, f.w);
  ((uint2*)dst)[i] = pk;
}

// ---------------------------------------------------------------------------
// Kernel 0b: fp32 W[k][n] -> bf16 W^T[n][k]  (coalesced read, strided write)
// ---------------------------------------------------------------------------
__global__ void convert_wT_kernel(const float* __restrict__ src,
                                  unsigned short* __restrict__ dst) {
  int i = blockIdx.x * 256 + threadIdx.x;     // i = k*HID + n
  int k = i >> 10;
  int n = i & (HID_ - 1);
  dst[(size_t)n * HID_ + k] = f2bf(src[i]);
}

// ---------------------------------------------------------------------------
// Kernel 1: C = X * W + bias (scaled). X bf16 [m][k], W^T bf16 [n][k].
// Block = 128 threads (4 waves), 64x64 tile, double-buffered async staging.
// vtrans=1 stores [B,NH,HD,S].
// ---------------------------------------------------------------------------
__global__ __launch_bounds__(128) void qkv_gemm_kernel(
    const unsigned short* __restrict__ Xb,   // [B*S, HID] bf16
    const unsigned short* __restrict__ WbT,  // [HID(n)][HID(k)] bf16
    const float* __restrict__ bias,          // [HID]
    unsigned short* __restrict__ dst,
    float scale, int vtrans) {
  __shared__ __align__(16) unsigned short As[2][64 * 32];  // [m][k]
  __shared__ __align__(16) unsigned short Bt[2][64 * 32];  // [n][k]

  const int tid  = threadIdx.x;
  const int lane = tid & 31;
  const int wv   = tid >> 5;
  const int m0   = blockIdx.x * 64;
  const int n0   = blockIdx.y * 64;

  auto stage = [&](int buf, int k0) {
    #pragma unroll
    for (int p = 0; p < 2; ++p) {
      int lin = p * 128 + tid;                // 256 b128 chunks per tile
      int row = lin >> 2;                     // 0..63
      int c8  = (lin & 3) << 3;               // k offset 0/8/16/24
      async_copy_b128(&As[buf][row * 32 + c8],
                      Xb + (size_t)(m0 + row) * HID_ + k0 + c8);
      async_copy_b128(&Bt[buf][row * 32 + c8],
                      WbT + (size_t)(n0 + row) * HID_ + k0 + c8);
    }
  };

  v8f acc[4] = {};

  stage(0, 0);
  wait_async0();
  __syncthreads();

  for (int k0 = 0; k0 < HID_; k0 += 32) {
    const int buf = (k0 >> 5) & 1;
    if (k0 + 32 < HID_) stage(buf ^ 1, k0 + 32);   // DMA next tile now

    v16bf a = lds_frag(As[buf], wv * 16, 32, 0);
    v16bf bfr0 = lds_frag(Bt[buf],  0, 32, 0);
    v16bf bfr1 = lds_frag(Bt[buf], 16, 32, 0);
    v16bf bfr2 = lds_frag(Bt[buf], 32, 32, 0);
    v16bf bfr3 = lds_frag(Bt[buf], 48, 32, 0);
    acc[0] = __builtin_amdgcn_wmma_f32_16x16x32_bf16(
        false, a, false, bfr0, (short)0, acc[0], false, false);
    acc[1] = __builtin_amdgcn_wmma_f32_16x16x32_bf16(
        false, a, false, bfr1, (short)0, acc[1], false, false);
    acc[2] = __builtin_amdgcn_wmma_f32_16x16x32_bf16(
        false, a, false, bfr2, (short)0, acc[2], false, false);
    acc[3] = __builtin_amdgcn_wmma_f32_16x16x32_bf16(
        false, a, false, bfr3, (short)0, acc[3], false, false);

    wait_async0();                            // next tile DMA complete
    __syncthreads();                          // all waves done with buf
  }

  // Epilogue: bias + scale, store bf16 (head-major; V transposed per-head)
  const int rowHalf = (lane >> 4) << 3;       // C-layout: lanes>=16 = rows 8..15
  #pragma unroll
  for (int t = 0; t < 4; ++t) {
    int n  = n0 + t * 16 + (lane & 15);
    float bvv = bias[n];
    int h = n >> 6;
    int d = n & 63;
    #pragma unroll
    for (int r = 0; r < 8; ++r) {
      int m = m0 + wv * 16 + rowHalf + r;
      int b = m >> 11;                        // / S_
      int s = m & (S_ - 1);
      float v = (acc[t][r] + bvv) * scale;
      size_t idx = vtrans
          ? ((size_t)(b * NH_ + h) * HD_ + d) * S_ + s    // [B,NH,HD,S]
          : ((size_t)(b * NH_ + h) * S_ + s) * HD_ + d;   // [B,NH,S,HD]
      dst[idx] = f2bf(v);
    }
  }
}

// ---------------------------------------------------------------------------
// Kernel 2: flash attention. Block = 128 threads (4 waves) per (b, h, 64 rows).
// Double-buffered async K/V staging; V already d-major in memory.
// ---------------------------------------------------------------------------
__global__ __launch_bounds__(128) void attn_kernel(
    const unsigned short* __restrict__ Q,   // [B,NH,S,HD] bf16 (pre-scaled)
    const unsigned short* __restrict__ K,   // [B,NH,S,HD]
    const unsigned short* __restrict__ V,   // [B,NH,HD,S] (transposed)
    const float* __restrict__ mask,         // [B,S]
    float* __restrict__ out) {              // [B,S,HID] fp32
  __shared__ __align__(16) unsigned short Qs[64 * 64];
  __shared__ __align__(16) unsigned short Ks[2][64 * 64];    // [key][d]
  __shared__ __align__(16) unsigned short Vt[2][64 * 64];    // [d][key]
  __shared__ __align__(16) unsigned short Ps[4][16 * 64];    // per-wave [m][key]

  const int tid  = threadIdx.x;
  const int lane = tid & 31;
  const int wv   = tid >> 5;
  const int q0   = blockIdx.x * 64;
  const int h    = blockIdx.y;
  const int b    = blockIdx.z;
  const size_t headBase = (size_t)(b * NH_ + h) * S_ * HD_;  // same for V

  auto stageKV = [&](int buf, int j0) {
    #pragma unroll
    for (int p = 0; p < 4; ++p) {
      int lin = p * 128 + tid;
      int row = lin >> 3;
      int c8  = (lin & 7) << 3;
      async_copy_b128(&Ks[buf][row * 64 + c8],
                      K + headBase + (size_t)(j0 + row) * HD_ + c8);
      async_copy_b128(&Vt[buf][row * 64 + c8],
                      V + headBase + (size_t)row * S_ + j0 + c8);
    }
  };

  // Stage Q tile (async), pull per-wave fragments into registers.
  #pragma unroll
  for (int p = 0; p < 4; ++p) {
    int lin = p * 128 + tid;
    int row = lin >> 3;
    int c8  = (lin & 7) << 3;
    async_copy_b128(&Qs[row * 64 + c8],
                    Q + headBase + (size_t)(q0 + row) * HD_ + c8);
  }
  stageKV(0, 0);                              // first K/V tile in same batch
  wait_async0();
  __syncthreads();
  v16bf qf0 = lds_frag(Qs, wv * 16, 64, 0);
  v16bf qf1 = lds_frag(Qs, wv * 16, 64, 32);

  float rowMax[8], rowSum[8];
  #pragma unroll
  for (int r = 0; r < 8; ++r) { rowMax[r] = -3.0e38f; rowSum[r] = 0.0f; }
  v8f o[4] = {};

  for (int j0 = 0; j0 < S_; j0 += 64) {
    const int buf = (j0 >> 6) & 1;
    if (j0 + 64 < S_) stageKV(buf ^ 1, j0 + 64);   // DMA next tile now

    // Scores: S = Q * K^T  (16x64 per wave), Q pre-scaled by 1/sqrt(HD)
    v8f sc[4] = {};
    #pragma unroll
    for (int t = 0; t < 4; ++t) {
      v16bf kf0 = lds_frag(Ks[buf], t * 16, 64, 0);
      v16bf kf1 = lds_frag(Ks[buf], t * 16, 64, 32);
      sc[t] = __builtin_amdgcn_wmma_f32_16x16x32_bf16(
          false, qf0, false, kf0, (short)0, sc[t], false, false);
      sc[t] = __builtin_amdgcn_wmma_f32_16x16x32_bf16(
          false, qf1, false, kf1, (short)0, sc[t], false, false);
    }
    #pragma unroll
    for (int t = 0; t < 4; ++t) {
      float mv = mask[b * S_ + j0 + t * 16 + (lane & 15)];
      #pragma unroll
      for (int r = 0; r < 8; ++r) sc[t][r] += mv;
    }

    // Online softmax. C-layout: each lane holds 8 rows of one column, so
    // row reductions are shfl_xor over the 16-lane half.
    unsigned short* P = Ps[wv];
    const int rowHalf = (lane >> 4) << 3;
    #pragma unroll
    for (int r = 0; r < 8; ++r) {
      float m = fmaxf(fmaxf(sc[0][r], sc[1][r]), fmaxf(sc[2][r], sc[3][r]));
      #pragma unroll
      for (int x = 1; x < 16; x <<= 1) m = fmaxf(m, __shfl_xor(m, x, 32));
      float mNew  = fmaxf(rowMax[r], m);
      float alpha = __expf(rowMax[r] - mNew);
      rowMax[r] = mNew;
      float ps = 0.0f;
      #pragma unroll
      for (int t = 0; t < 4; ++t) {
        float p = __expf(sc[t][r] - mNew);
        sc[t][r] = p;
        ps += p;
      }
      #pragma unroll
      for (int x = 1; x < 16; x <<= 1) ps += __shfl_xor(ps, x, 32);
      rowSum[r] = rowSum[r] * alpha + ps;
      #pragma unroll
      for (int t = 0; t < 4; ++t) o[t][r] *= alpha;
    }
    // Re-layout P (C-frag -> A-frag) via per-wave LDS tile; convert in pairs.
    #pragma unroll
    for (int t = 0; t < 4; ++t) {
      int col = t * 16 + (lane & 15);
      #pragma unroll
      for (int r = 0; r < 8; r += 2) {
        unsigned pk = pack2bf(sc[t][r], sc[t][r + 1]);
        P[(rowHalf + r)     * 64 + col] = (unsigned short)pk;
        P[(rowHalf + r + 1) * 64 + col] = (unsigned short)(pk >> 16);
      }
    }
    __syncthreads();

    // O += P * V  (contraction over keys; d-major V gives contiguous B-frags)
    v16bf pf0 = lds_frag(P, 0, 64, 0);
    v16bf pf1 = lds_frag(P, 0, 64, 32);
    #pragma unroll
    for (int t = 0; t < 4; ++t) {
      v16bf vf0 = lds_frag(Vt[buf], t * 16, 64, 0);
      v16bf vf1 = lds_frag(Vt[buf], t * 16, 64, 32);
      o[t] = __builtin_amdgcn_wmma_f32_16x16x32_bf16(
          false, pf0, false, vf0, (short)0, o[t], false, false);
      o[t] = __builtin_amdgcn_wmma_f32_16x16x32_bf16(
          false, pf1, false, vf1, (short)0, o[t], false, false);
    }

    wait_async0();                            // next K/V tile DMA complete
    __syncthreads();                          // all waves done with buf
  }

  // Epilogue: normalize and store fp32 [B, S, HID]
  const int rowHalf = (lane >> 4) << 3;
  #pragma unroll
  for (int t = 0; t < 4; ++t) {
    int d = t * 16 + (lane & 15);
    #pragma unroll
    for (int r = 0; r < 8; ++r) {
      int s = q0 + wv * 16 + rowHalf + r;
      out[((size_t)b * S_ + s) * HID_ + h * HD_ + d] = o[t][r] / rowSum[r];
    }
  }
}

// ---------------------------------------------------------------------------
// Launch
// ---------------------------------------------------------------------------
extern "C" void kernel_launch(void* const* d_in, const int* in_sizes, int n_in,
                              void* d_out, int out_size, void* d_ws, size_t ws_size,
                              hipStream_t stream) {
  const float* X    = (const float*)d_in[0];  // hidden_states [B,S,HID]
  const float* mask = (const float*)d_in[1];  // [B,1,1,S]
  const float* Wq   = (const float*)d_in[2];
  const float* bq   = (const float*)d_in[3];
  const float* Wk   = (const float*)d_in[4];
  const float* bk   = (const float*)d_in[5];
  const float* Wv   = (const float*)d_in[6];
  const float* bv   = (const float*)d_in[7];
  float* out = (float*)d_out;

  char* ws = (char*)d_ws;
  // Workspace: 3x2MB W^T | 3x16MB Q/K/V | 16MB X(bf16)  => 70MB total.
  unsigned short* WqT = (unsigned short*)(ws);
  unsigned short* WkT = (unsigned short*)(ws + ((size_t)2  << 20));
  unsigned short* WvT = (unsigned short*)(ws + ((size_t)4  << 20));
  unsigned short* Qw  = (unsigned short*)(ws + ((size_t)6  << 20));
  unsigned short* Kw  = (unsigned short*)(ws + ((size_t)22 << 20));
  unsigned short* Vw  = (unsigned short*)(ws + ((size_t)38 << 20));
  unsigned short* Xb  = (unsigned short*)(ws + ((size_t)54 << 20));

  const int wn = HID_ * HID_;
  const int xn4 = (B_ * S_ * HID_) / 4;
  convert_x_kernel <<<xn4 / 256, 256, 0, stream>>>(X, Xb);
  convert_wT_kernel<<<wn / 256, 256, 0, stream>>>(Wq, WqT);
  convert_wT_kernel<<<wn / 256, 256, 0, stream>>>(Wk, WkT);
  convert_wT_kernel<<<wn / 256, 256, 0, stream>>>(Wv, WvT);

  dim3 gemmGrid((B_ * S_) / 64, HID_ / 64);   // 128 x 16
  qkv_gemm_kernel<<<gemmGrid, 128, 0, stream>>>(Xb, WqT, bq, Qw, 0.125f, 0);
  qkv_gemm_kernel<<<gemmGrid, 128, 0, stream>>>(Xb, WkT, bk, Kw, 1.0f,   0);
  qkv_gemm_kernel<<<gemmGrid, 128, 0, stream>>>(Xb, WvT, bv, Vw, 1.0f,   1);

  dim3 attnGrid(S_ / 64, NH_, B_);            // 32 x 16 x 4
  attn_kernel<<<attnGrid, 128, 0, stream>>>(Qw, Kw, Vw, mask, out);
}
